// AFF_74861279969483
// MI455X (gfx1250) — compile-verified, matrix operand order checked by reading, not run on hardware
//
#include <hip/hip_runtime.h>
#include <math.h>

typedef __attribute__((ext_vector_type(16))) _Float16 v16h;
typedef __attribute__((ext_vector_type(8)))  float    v8f;

#define TILE_M 128
#define TILE_N 128
#define TILE_K 32

#if __has_builtin(__builtin_amdgcn_global_load_async_to_lds_b128) && \
    __has_builtin(__builtin_amdgcn_s_wait_asynccnt)
#define ATTN_ASYNC_LDS 1
// Exact builtin parameter types (from clang's diagnostic): pointer to
// int32x4 (GCC vector_size spelling) in AS1 (global) / AS3 (LDS).
typedef int async_v4i __attribute__((vector_size(16)));
typedef __attribute__((address_space(1))) async_v4i* async_gptr;
typedef __attribute__((address_space(3))) async_v4i* async_lptr;
#else
#define ATTN_ASYNC_LDS 0
#endif

// ---------------------------------------------------------------------------
// Generic 128x128x32 double-buffered WMMA GEMM:  C = A[M,K] @ B[N,K]^T (+bias)
// Both operands are K-contiguous (row-major with K innermost), which matches
// the wave32 WMMA fragment feed: each lane loads 32 contiguous bytes.
// AF32/BF32: operand in f32 (converted to f16 during LDS staging) or f16.
// f16 x f16 GEMMs use GLOBAL_LOAD_ASYNC_TO_LDS (ASYNCcnt-tracked DMA) when
// the toolchain exposes the gfx1250 builtins; otherwise reg staging.
// EPI: 0 = f16 store (+opt bias)        [projections q/k, attn@v]
//      1 = f16 transposed store (vT)    [v projection -> vT[b,d,n]]
//      2 = f32 store, no bias           [energy]
//      3 = fused gate: out*sigmoid(s)*gamma + x1 -> f32 d_out
// ---------------------------------------------------------------------------
template<int AF32, int BF32, int EPI>
__global__ __launch_bounds__(256)
void wmma_gemm(const void* __restrict__ Ap, const void* __restrict__ Bp,
               const float* __restrict__ bias, void* __restrict__ Cp,
               int M, int Nout, int K,
               long long sA, long long sB, long long sC, int vtN,
               const _Float16* __restrict__ gOut, const float* __restrict__ gX1,
               const float* __restrict__ gGamma, float* __restrict__ gFinal)
{
    __shared__ _Float16 smA[2][TILE_M * TILE_K];
    __shared__ _Float16 smB[2][TILE_N * TILE_K];

    const int tid   = threadIdx.x;
    const int lane  = tid & 31;
    const int wave  = tid >> 5;
    const int wm    = wave >> 2;          // 0..1 : 64-row slab
    const int wn    = wave & 3;           // 0..3 : 32-col slab
    const int laneR = lane & 15;
    const int laneK = (lane >> 4) * 16;
    const size_t bz = blockIdx.z;

    const float*    Af = (const float*)Ap    + bz * (size_t)sA;
    const _Float16* Ah = (const _Float16*)Ap + bz * (size_t)sA;
    const float*    Bf = (const float*)Bp    + bz * (size_t)sB;
    const _Float16* Bh = (const _Float16*)Bp + bz * (size_t)sB;

    // staging: 256 threads cover 128 rows x 32 K, 16 elems (half-row) each
    const int srow = tid >> 1;
    const int sseg = (tid & 1) * 16;
    const size_t aRow = (size_t)blockIdx.y * TILE_M + srow;
    const size_t bRow = (size_t)blockIdx.x * TILE_N + srow;

    float4 ra[4], rb[4];

    auto loadG = [&](int kk) {
        if (AF32) {
            const float4* g = (const float4*)(Af + aRow * (size_t)K + kk + sseg);
            ra[0] = g[0]; ra[1] = g[1]; ra[2] = g[2]; ra[3] = g[3];
        } else {
            const float4* g = (const float4*)(Ah + aRow * (size_t)K + kk + sseg);
            ra[0] = g[0]; ra[1] = g[1];
        }
        if (BF32) {
            const float4* g = (const float4*)(Bf + bRow * (size_t)K + kk + sseg);
            rb[0] = g[0]; rb[1] = g[1]; rb[2] = g[2]; rb[3] = g[3];
        } else {
            const float4* g = (const float4*)(Bh + bRow * (size_t)K + kk + sseg);
            rb[0] = g[0]; rb[1] = g[1];
        }
    };
    auto storeL = [&](int buf) {
        _Float16* da = &smA[buf][srow * TILE_K + sseg];
        if (AF32) {
            const float* f = (const float*)ra;
            _Float16 t[16];
            #pragma unroll
            for (int i = 0; i < 16; ++i) t[i] = (_Float16)f[i];
            ((float4*)da)[0] = ((const float4*)t)[0];
            ((float4*)da)[1] = ((const float4*)t)[1];
        } else {
            ((float4*)da)[0] = ra[0];
            ((float4*)da)[1] = ra[1];
        }
        _Float16* db = &smB[buf][srow * TILE_K + sseg];
        if (BF32) {
            const float* f = (const float*)rb;
            _Float16 t[16];
            #pragma unroll
            for (int i = 0; i < 16; ++i) t[i] = (_Float16)f[i];
            ((float4*)db)[0] = ((const float4*)t)[0];
            ((float4*)db)[1] = ((const float4*)t)[1];
        } else {
            ((float4*)db)[0] = rb[0];
            ((float4*)db)[1] = rb[1];
        }
    };

    v8f acc[4][2] = {};

    auto computeStep = [&](int cur) {
        const _Float16* sa = &smA[cur][(wm * 64 + laneR) * TILE_K + laneK];
        const _Float16* sb = &smB[cur][(wn * 32 + laneR) * TILE_K + laneK];
        v16h a0 = *(const v16h*)(sa + 0 * 16 * TILE_K);
        v16h a1 = *(const v16h*)(sa + 1 * 16 * TILE_K);
        v16h a2 = *(const v16h*)(sa + 2 * 16 * TILE_K);
        v16h a3 = *(const v16h*)(sa + 3 * 16 * TILE_K);
        v16h b0 = *(const v16h*)(sb + 0 * 16 * TILE_K);
        v16h b1 = *(const v16h*)(sb + 1 * 16 * TILE_K);
        acc[0][0] = __builtin_amdgcn_wmma_f32_16x16x32_f16(false, a0, false, b0, (short)0, acc[0][0], false, false);
        acc[0][1] = __builtin_amdgcn_wmma_f32_16x16x32_f16(false, a0, false, b1, (short)0, acc[0][1], false, false);
        acc[1][0] = __builtin_amdgcn_wmma_f32_16x16x32_f16(false, a1, false, b0, (short)0, acc[1][0], false, false);
        acc[1][1] = __builtin_amdgcn_wmma_f32_16x16x32_f16(false, a1, false, b1, (short)0, acc[1][1], false, false);
        acc[2][0] = __builtin_amdgcn_wmma_f32_16x16x32_f16(false, a2, false, b0, (short)0, acc[2][0], false, false);
        acc[2][1] = __builtin_amdgcn_wmma_f32_16x16x32_f16(false, a2, false, b1, (short)0, acc[2][1], false, false);
        acc[3][0] = __builtin_amdgcn_wmma_f32_16x16x32_f16(false, a3, false, b0, (short)0, acc[3][0], false, false);
        acc[3][1] = __builtin_amdgcn_wmma_f32_16x16x32_f16(false, a3, false, b1, (short)0, acc[3][1], false, false);
    };

    const int nk = K / TILE_K;

#if ATTN_ASYNC_LDS
    if (!AF32 && !BF32) {
        // ASYNCcnt-tracked DMA straight into LDS: no staging VGPRs, DMA
        // overlaps with WMMA across the double buffer. INST_OFFSET applies to
        // both LDS and global addresses, so two b128 issues cover 32B/lane.
        auto stageAsync = [&](int buf, int kk) {
            const _Float16* ga = Ah + aRow * (size_t)K + kk + sseg;
            _Float16*       la = &smA[buf][srow * TILE_K + sseg];
            __builtin_amdgcn_global_load_async_to_lds_b128(
                (async_gptr)ga, (async_lptr)la, 0, 0);
            __builtin_amdgcn_global_load_async_to_lds_b128(
                (async_gptr)ga, (async_lptr)la, 16, 0);
            const _Float16* gb = Bh + bRow * (size_t)K + kk + sseg;
            _Float16*       lb = &smB[buf][srow * TILE_K + sseg];
            __builtin_amdgcn_global_load_async_to_lds_b128(
                (async_gptr)gb, (async_lptr)lb, 0, 0);
            __builtin_amdgcn_global_load_async_to_lds_b128(
                (async_gptr)gb, (async_lptr)lb, 16, 0);
        };
        stageAsync(0, 0);
        __builtin_amdgcn_s_wait_asynccnt(0);
        __syncthreads();
        for (int kt = 0; kt < nk; ++kt) {
            const int cur = kt & 1;
            if (kt + 1 < nk) stageAsync(cur ^ 1, (kt + 1) * TILE_K);
            computeStep(cur);
            __builtin_amdgcn_s_wait_asynccnt(0);
            __syncthreads();
        }
    } else
#endif
    {
        loadG(0);
        storeL(0);
        __syncthreads();
        for (int kt = 0; kt < nk; ++kt) {
            const int cur = kt & 1;
            if (kt + 1 < nk) loadG((kt + 1) * TILE_K);
            if (kt + 2 < nk) {   // gfx1250 global_prefetch_b8 two tiles ahead
                if (AF32) __builtin_prefetch(Af + aRow * (size_t)K + (kt + 2) * TILE_K + sseg, 0, 1);
                else      __builtin_prefetch(Ah + aRow * (size_t)K + (kt + 2) * TILE_K + sseg, 0, 1);
                if (BF32) __builtin_prefetch(Bf + bRow * (size_t)K + (kt + 2) * TILE_K + sseg, 0, 1);
                else      __builtin_prefetch(Bh + bRow * (size_t)K + (kt + 2) * TILE_K + sseg, 0, 1);
            }
            computeStep(cur);
            if (kt + 1 < nk) storeL(cur ^ 1);
            __syncthreads();
        }
    }

    // Epilogue. C/D fragment layout: VGPR v, lanes 0-15 -> row v, lanes 16-31
    // -> row v+8; column = lane & 15.
    const int rowBase = blockIdx.y * TILE_M + wm * 64;
    const int colBase = blockIdx.x * TILE_N + wn * 32;
    const float gval = (EPI == 3) ? gGamma[0] : 0.0f;

    #pragma unroll
    for (int fm = 0; fm < 4; ++fm) {
        #pragma unroll
        for (int fn = 0; fn < 2; ++fn) {
            const int col = colBase + fn * 16 + laneR;
            float bvv = 0.0f;
            if (EPI != 2 && bias != nullptr) bvv = bias[col];
            #pragma unroll
            for (int v = 0; v < 8; ++v) {
                const int row = rowBase + fm * 16 + (lane >> 4) * 8 + v;
                float val = acc[fm][fn][v] + bvv;
                if (EPI == 0) {
                    _Float16* C = (_Float16*)Cp + bz * (size_t)sC;
                    C[(size_t)row * Nout + col] = (_Float16)val;
                } else if (EPI == 1) {
                    _Float16* C = (_Float16*)Cp;   // vT[b, col, n]
                    const int bb = row / vtN;
                    const int nn = row % vtN;
                    C[((size_t)bb * Nout + col) * (size_t)vtN + nn] = (_Float16)val;
                } else if (EPI == 2) {
                    float* C = (float*)Cp + bz * (size_t)sC;
                    C[(size_t)row * Nout + col] = val;
                } else {
                    const size_t idx = (size_t)row * Nout + col;
                    const float o  = (float)gOut[idx];
                    const float sg = 1.0f / (1.0f + __expf(-val));
                    gFinal[idx] = gval * o * sg + gX1[idx];
                }
            }
        }
    }
}

// ---------------------------------------------------------------------------
// Row softmax: one 256-thread block per row of 2048 f32 energies -> f16 attn
// ---------------------------------------------------------------------------
__global__ __launch_bounds__(256)
void softmax_rows(const float* __restrict__ E, _Float16* __restrict__ A, int ncols)
{
    const size_t row = blockIdx.x;
    const float* e = E + row * (size_t)ncols;
    _Float16*    a = A + row * (size_t)ncols;
    const int tid = threadIdx.x;

    float v[8];
    float m = -INFINITY;
    #pragma unroll
    for (int i = 0; i < 8; ++i) { v[i] = e[tid + i * 256]; m = fmaxf(m, v[i]); }

    __shared__ float red[256];
    red[tid] = m; __syncthreads();
    #pragma unroll
    for (int s = 128; s > 0; s >>= 1) {
        if (tid < s) red[tid] = fmaxf(red[tid], red[tid + s]);
        __syncthreads();
    }
    m = red[0];
    __syncthreads();

    float sum = 0.0f;
    #pragma unroll
    for (int i = 0; i < 8; ++i) { v[i] = __expf(v[i] - m); sum += v[i]; }
    red[tid] = sum; __syncthreads();
    #pragma unroll
    for (int s = 128; s > 0; s >>= 1) {
        if (tid < s) red[tid] += red[tid + s];
        __syncthreads();
    }
    const float inv = 1.0f / red[0];
    #pragma unroll
    for (int i = 0; i < 8; ++i) a[tid + i * 256] = (_Float16)(v[i] * inv);
}

// ---------------------------------------------------------------------------
// Combine gate weights: Wc = f16(W1+W2+W3), bc = b1+b2+b3
// ---------------------------------------------------------------------------
__global__ __launch_bounds__(256)
void prep_gate(const float* __restrict__ W1, const float* __restrict__ W2,
               const float* __restrict__ W3, const float* __restrict__ b1,
               const float* __restrict__ b2, const float* __restrict__ b3,
               _Float16* __restrict__ Wc, float* __restrict__ bc, int DD, int D)
{
    const int i = blockIdx.x * 256 + threadIdx.x;
    if (i < DD) Wc[i] = (_Float16)(W1[i] + W2[i] + W3[i]);
    if (i < D)  bc[i] = b1[i] + b2[i] + b3[i];
}

// ---------------------------------------------------------------------------
extern "C" void kernel_launch(void* const* d_in, const int* in_sizes, int n_in,
                              void* d_out, int out_size, void* d_ws, size_t ws_size,
                              hipStream_t stream)
{
    const int Bb = 16, Nn = 2048, Dd = 1024;
    const int Mrows = Bb * Nn;                    // 32768

    const float* x1 = (const float*)d_in[0];
    const float* x2 = (const float*)d_in[1];
    const float* Wq = (const float*)d_in[2];
    const float* bq = (const float*)d_in[3];
    const float* Wk = (const float*)d_in[4];
    const float* bk = (const float*)d_in[5];
    const float* Wv = (const float*)d_in[6];
    const float* bv = (const float*)d_in[7];
    const float* W1 = (const float*)d_in[8];
    const float* b1 = (const float*)d_in[9];
    const float* W2 = (const float*)d_in[10];
    const float* b2 = (const float*)d_in[11];
    const float* W3 = (const float*)d_in[12];
    const float* b3 = (const float*)d_in[13];
    const float* gamma = (const float*)d_in[14];

    char* ws = (char*)d_ws;
    size_t off = 0;
    auto carve = [&](size_t bytes) -> char* {
        char* p = ws + off;
        off += (bytes + 255) & ~(size_t)255;
        return p;
    };
    const size_t QE = (size_t)Mrows * Dd;         // q/k/v/out element count
    const size_t AE = (size_t)Bb * Nn * Nn;       // attention matrix elements

    _Float16* q      = (_Float16*)carve(QE * 2);
    _Float16* kmat   = (_Float16*)carve(QE * 2);
    _Float16* vT     = (_Float16*)carve(QE * 2);
    _Float16* outH   = (_Float16*)carve(QE * 2);
    _Float16* attn   = (_Float16*)carve(AE * 2);
    float*    energy = (float*)carve(AE * 4);
    _Float16* Wc     = (_Float16*)carve((size_t)Dd * Dd * 2);
    float*    bc     = (float*)carve((size_t)Dd * 4);

    dim3 blk(256);

    // 0) combine gate weights (tiny)
    prep_gate<<<dim3((Dd * Dd + 255) / 256), blk, 0, stream>>>(
        W1, W2, W3, b1, b2, b3, Wc, bc, Dd * Dd, Dd);

    // 1) projections: q = x1@Wq^T+bq, k = x2@Wk^T+bk, vT = (x2@Wv^T+bv)^T
    wmma_gemm<1, 1, 0><<<dim3(Dd / 128, Mrows / 128, 1), blk, 0, stream>>>(
        x1, Wq, bq, q, Mrows, Dd, Dd, 0, 0, 0, Nn, nullptr, nullptr, nullptr, nullptr);
    wmma_gemm<1, 1, 0><<<dim3(Dd / 128, Mrows / 128, 1), blk, 0, stream>>>(
        x2, Wk, bk, kmat, Mrows, Dd, Dd, 0, 0, 0, Nn, nullptr, nullptr, nullptr, nullptr);
    wmma_gemm<1, 1, 1><<<dim3(Dd / 128, Mrows / 128, 1), blk, 0, stream>>>(
        x2, Wv, bv, vT, Mrows, Dd, Dd, 0, 0, 0, Nn, nullptr, nullptr, nullptr, nullptr);

    // 2) energy[b] = q[b] @ k[b]^T   (f32 out)
    wmma_gemm<0, 0, 2><<<dim3(Nn / 128, Nn / 128, Bb), blk, 0, stream>>>(
        q, kmat, nullptr, energy, Nn, Nn, Dd,
        (long long)Nn * Dd, (long long)Nn * Dd, (long long)Nn * Nn, Nn,
        nullptr, nullptr, nullptr, nullptr);

    // 3) softmax rows -> f16 attn
    softmax_rows<<<dim3(Bb * Nn), blk, 0, stream>>>(energy, attn, Nn);

    // 4) out[b] = attn[b] @ v[b]   (vT supplies K-contiguous B operand)
    wmma_gemm<0, 0, 0><<<dim3(Dd / 128, Nn / 128, Bb), blk, 0, stream>>>(
        attn, vT, nullptr, outH, Nn, Dd, Nn,
        (long long)Nn * Nn, (long long)Dd * Nn, (long long)Nn * Dd, Nn,
        nullptr, nullptr, nullptr, nullptr);

    // 5) s = out@Wc^T + bc; d_out = gamma * out * sigmoid(s) + x1   (fused)
    wmma_gemm<0, 0, 3><<<dim3(Dd / 128, Mrows / 128, 1), blk, 0, stream>>>(
        outH, Wc, bc, nullptr, Mrows, Dd, Dd, 0, 0, 0, Nn,
        outH, x1, gamma, (float*)d_out);
}
